// rnn_pim_layer_49924699849113
// MI455X (gfx1250) — matrix operand order repaired and checked by reading, not compiled, not run on hardware
//
#include <hip/hip_runtime.h>

// ---------------------------------------------------------------------------
// Multi-layer bidirectional LSTM for MI455X (gfx1250), bf16 WMMA datapath.
//   T=237, B=32, IN0=2592, H=1024, L=2, BI=2
//   * weights converted once to bf16, transposed to [outCol][row] (Wt)
//   * input-projection GEMMs: LDS triple-buffered stages filled with
//     global_load_async_to_lds_* (ASYNCcnt pipeline, 2 stages in flight),
//     branch-free steady-state loop, WMMA feeds from LDS via ds_load_b128
//   * recurrent scan: persistent kernel, 8 blocks/direction, B fragments
//     shared across both M-tiles and register-prefetched one k-step ahead
// ---------------------------------------------------------------------------

#define T_SEQ 237
#define NBAT  32
#define IN0   2592
#define HID   1024
#define G4    4096      // 4*H
#define WYC   8192      // BI*4*H  (weight columns)
#define WXR   6688      // flat weight rows
#define NBLK  8         // blocks per direction in the scan
#define KS    (HID / NBLK)

#define BKG   32                    // GEMM k-stage
#define ABUFH (32 * BKG)            // A tile halves per stage (1024)
#define STAGEH (ABUFH + 256 * BKG)  // halves per stage (9216 -> 18KB)

typedef __attribute__((ext_vector_type(16))) __bf16        bf16x16;
typedef __attribute__((ext_vector_type(8)))  float         f32x8;
typedef __attribute__((ext_vector_type(4)))  unsigned int  u32x4;

union FragCvt { u32x4 u[2]; bf16x16 v; };

__device__ __forceinline__ unsigned short f2bf(float x) {
  unsigned u = __builtin_bit_cast(unsigned, x);
  u += 0x7FFFu + ((u >> 16) & 1u);          // round to nearest even
  return (unsigned short)(u >> 16);
}

__device__ __forceinline__ float sigm(float x)      { return 1.0f / (1.0f + __expf(-x)); }
__device__ __forceinline__ float tanh_fast(float x) { float e = __expf(2.0f * x); return 1.0f - 2.0f / (e + 1.0f); }

// A-matrix 16x32 bf16 fragment (ISA 7.12.2): lanes 0-15 row M=lane, K {k0..+7, k0+16..+23};
// lanes 16-31 row M=lane-16, K {k0+8..+15, k0+24..+31}.
__device__ __forceinline__ bf16x16 load_a_frag(const unsigned short* base, int ld, int row0, int k0) {
  int lane = threadIdx.x & 31;
  int half = lane >> 4;
  int r    = lane & 15;
  const unsigned short* p = base + (size_t)(row0 + r) * ld + k0 + half * 8;
  FragCvt f;
  f.u[0] = *(const u32x4*)(p);
  f.u[1] = *(const u32x4*)(p + 16);
  return f.v;
}

// B-matrix 32x16 bf16 fragment, column-major per lane: lanes 0-15 col N=lane hold
// K {k0..+7, k0+16..+23}; lanes 16-31 col N=lane-16 hold K {k0+8..+15, k0+24..+31}. Wt is [N][K].
__device__ __forceinline__ bf16x16 load_b_frag(const unsigned short* base, int ld, int col0, int k0) {
  int lane = threadIdx.x & 31;
  int half = lane >> 4;
  int r    = lane & 15;
  const unsigned short* p = base + (size_t)(col0 + r) * ld + k0 + half * 8;
  FragCvt f;
  f.u[0] = *(const u32x4*)(p);
  f.u[1] = *(const u32x4*)(p + 16);
  return f.v;
}

__device__ __forceinline__ f32x8 wmma_bf16(bf16x16 a, bf16x16 b, f32x8 c) {
  return __builtin_amdgcn_wmma_f32_16x16x32_bf16(false, a, false, b, (short)0, c, false, false);
}

// --------------------------- helper kernels --------------------------------

__global__ void transpose_cvt_w(const float* __restrict__ W, unsigned short* __restrict__ Wt) {
  int k = blockIdx.x * 32 + threadIdx.x;   // 0..WXR-1
  int n = blockIdx.y * 8  + threadIdx.y;   // 0..WYC-1
  Wt[(size_t)n * WXR + k] = f2bf(W[(size_t)k * WYC + n]);
}

__global__ void cvt_f32_bf16(const float* __restrict__ s, unsigned short* __restrict__ d, int n) {
  int i = blockIdx.x * 256 + threadIdx.x;
  if (i < n) d[i] = f2bf(s[i]);
}

__global__ void init_state(const float* __restrict__ h0, const float* __restrict__ c0,
                           unsigned short* __restrict__ hbf, float* __restrict__ cst,
                           unsigned* __restrict__ sync, int layer) {
  int idx = blockIdx.x * 256 + threadIdx.x;      // < 2*32*1024
  int d   = idx >> 15;
  int rem = idx & 32767;                          // b*1024 + k
  float hv = h0[(size_t)(layer * 2 + d) * 32768 + rem];
  float cv = c0[(size_t)(layer * 2 + d) * 32768 + rem];
  hbf[(size_t)(d * 2 + 0) * 32768 + rem] = f2bf(hv);
  cst[(size_t)d * 32768 + rem] = cv;
  if (idx < 64) sync[idx] = 0;
}

// ------------------- input-projection GEMM (xw = x @ Wd) -------------------
// C[M x 8192] = A[M x K](bf16) @ W[K x 8192] with W as Wt[8192 x WXR].
// Block tile 32x256, BK=32 stages staged into LDS via async-to-LDS copies,
// triple-buffered with 2 stages in flight; 5 async ops per wave per stage.
// Steady-state loop is branch-free (tail stages peeled, buffers rotated
// incrementally, per-thread global pointers bumped by BKG each stage).
__global__ void __launch_bounds__(256)
wmma_gemm_proj(const unsigned short* __restrict__ A, const unsigned short* __restrict__ Wt,
               float* __restrict__ C, int K, int k_off) {
  __shared__ unsigned short smem[3 * STAGEH];   // 54 KB

  int tid  = threadIdx.x;
  int wave = tid >> 5;
  int lane = tid & 31;
  int half = lane >> 4;
  int r    = lane & 15;
  int mt = wave & 1;
  int ng = wave >> 1;
  int m_blk = blockIdx.y * 32;
  int n_blk = blockIdx.x * 256;

  unsigned short* smem_esc = smem;   // escape LDS object into the asm so alias
                                     // analysis keeps the ds_load fragment reads

  // per-thread global source cursors, advanced by BKG halves per stage
  const unsigned short* gA = A + (size_t)(m_blk + (tid >> 3)) * K + (tid & 7) * 4;
  const unsigned short* gB = Wt + (size_t)(n_blk + tid) * WXR + k_off;
  const unsigned loffA_l = (unsigned)((tid >> 3) * BKG + (tid & 7) * 4) * 2u;
  const unsigned loffB_l = (unsigned)(ABUFH + tid * BKG) * 2u;

  // Issue one stage of async copies (global -> LDS). Per wave: 1 b64 (A) + 4 b128 (B).
  auto issue_stage = [&](int sbh) {
    unsigned sbase = (unsigned)sbh * 2u;          // LDS byte offset of this buffer
    asm volatile("global_load_async_to_lds_b64 %0, %1, off"
                 :: "v"(sbase + loffA_l), "v"(gA), "v"(smem_esc) : "memory");
#pragma unroll
    for (int i = 0; i < 4; ++i) {
      asm volatile("global_load_async_to_lds_b128 %0, %1, off"
                   :: "v"(sbase + loffB_l + 16u * i), "v"(gB + 8 * i), "v"(smem_esc) : "memory");
    }
    gA += BKG;
    gB += BKG;
  };

  f32x8 acc0 = {}, acc1 = {}, acc2 = {}, acc3 = {};
  const int ai_l = (mt * 16 + r) * BKG + half * 8;          // A frag LDS index
  const int bi_l = ABUFH + (ng * 64 + r) * BKG + half * 8;  // B frag LDS index (q adds 512)

  auto compute_stage = [&](int sbh) {
    FragCvt fa;
    fa.u[0] = *(const u32x4*)&smem[sbh + ai_l];
    fa.u[1] = *(const u32x4*)&smem[sbh + ai_l + 16];
#pragma unroll
    for (int q = 0; q < 4; ++q) {
      int bi = sbh + bi_l + q * 16 * BKG;
      FragCvt fb;
      fb.u[0] = *(const u32x4*)&smem[bi];
      fb.u[1] = *(const u32x4*)&smem[bi + 16];
      if (q == 0) acc0 = wmma_bf16(fa.v, fb.v, acc0);
      if (q == 1) acc1 = wmma_bf16(fa.v, fb.v, acc1);
      if (q == 2) acc2 = wmma_bf16(fa.v, fb.v, acc2);
      if (q == 3) acc3 = wmma_bf16(fa.v, fb.v, acc3);
    }
  };

  int nstage = K / BKG;          // 81 or 64, always >= 2
  issue_stage(0);
  issue_stage(STAGEH);

  int comp_sb  = 0;              // buffer being computed
  int issue_sb = 2 * STAGEH;     // buffer to fill next
  for (int st = 0; st < nstage - 2; ++st) {
    issue_stage(issue_sb);
    asm volatile("s_wait_asynccnt 10" ::: "memory");  // stage st landed (in-order, 5/stage)
    __syncthreads();
    compute_stage(comp_sb);
    __syncthreads();             // buffer free before it is refilled (stage st+3)
    issue_sb = comp_sb;          // consumed buffer becomes next fill target
    comp_sb  = (comp_sb == 2 * STAGEH) ? 0 : comp_sb + STAGEH;
  }
  // peeled tail: stages nstage-2 and nstage-1 (already in flight)
  asm volatile("s_wait_asynccnt 5" ::: "memory");
  __syncthreads();
  compute_stage(comp_sb);
  comp_sb = (comp_sb == 2 * STAGEH) ? 0 : comp_sb + STAGEH;
  asm volatile("s_wait_asynccnt 0" ::: "memory");
  __syncthreads();
  compute_stage(comp_sb);

  int mrow = m_blk + mt * 16 + 8 * half;
  int n0 = n_blk + ng * 64;
#pragma unroll
  for (int v = 0; v < 8; ++v) {
    size_t ro = (size_t)(mrow + v) * WYC;
    C[ro + n0 +  0 + r] = acc0[v];
    C[ro + n0 + 16 + r] = acc1[v];
    C[ro + n0 + 32 + r] = acc2[v];
    C[ro + n0 + 48 + r] = acc3[v];
  }
}

// ------------------------------ LSTM scan ----------------------------------
// grid = (NBLK, 2 directions). Block owns H-slice [kbase, kbase+KS).
// Each wave owns one 16-wide column slice (nt=wave) with BOTH m-tiles:
// B fragments loaded once per k-step (shared by both m-tiles) and
// register-prefetched one k-step ahead to overlap with the 8 WMMAs.
__global__ void __launch_bounds__(256)
lstm_scan(const unsigned short* __restrict__ Wt, const float* __restrict__ xw,
          float* __restrict__ out, unsigned short* __restrict__ hbf,
          float* __restrict__ cst, unsigned* __restrict__ sync, int r_off) {
  int tid  = threadIdx.x;
  int wave = tid >> 5, lane = tid & 31, half = lane >> 4, r = lane & 15;
  int dir   = blockIdx.y;
  int kbase = blockIdx.x * KS;
  int nt    = wave;                       // 8 waves x 16 cols = KS
  int ncol  = kbase + nt * 16;
  unsigned* syncd = sync + dir * 32;
  unsigned short* hb = hbf + (size_t)dir * 2 * 32768;   // ping/pong
  float* cd = cst + (size_t)dir * 32768;

  int cI = dir * G4 + 0 * HID + ncol;     // Wt column bases for the 4 gates
  int cF = dir * G4 + 1 * HID + ncol;
  int cG = dir * G4 + 2 * HID + ncol;
  int cO = dir * G4 + 3 * HID + ncol;

  for (int s = 0; s < T_SEQ; ++s) {
    int t = dir ? (T_SEQ - 1 - s) : s;
    const unsigned short* hread  = hb + (size_t)(s & 1) * 32768;
    unsigned short*       hwrite = hb + (size_t)((s + 1) & 1) * 32768;
    const float* xwt = xw + ((size_t)t * NBAT) * WYC + (size_t)dir * G4;

    // accumulators: 4 gates x 2 m-tiles, preloaded with x@W
    f32x8 aI0, aF0, aG0, aO0, aI1, aF1, aG1, aO1;
#pragma unroll
    for (int v = 0; v < 8; ++v) {
      size_t r0 = (size_t)(8 * half + v) * WYC;
      size_t r1 = (size_t)(16 + 8 * half + v) * WYC;
      aI0[v] = xwt[r0 + 0 * HID + ncol + r];  aI1[v] = xwt[r1 + 0 * HID + ncol + r];
      aF0[v] = xwt[r0 + 1 * HID + ncol + r];  aF1[v] = xwt[r1 + 1 * HID + ncol + r];
      aG0[v] = xwt[r0 + 2 * HID + ncol + r];  aG1[v] = xwt[r1 + 2 * HID + ncol + r];
      aO0[v] = xwt[r0 + 3 * HID + ncol + r];  aO1[v] = xwt[r1 + 3 * HID + ncol + r];
    }

    // software-pipelined k loop: prefetch next B fragments before WMMAs
    bf16x16 bI = load_b_frag(Wt, WXR, cI, r_off);
    bf16x16 bF = load_b_frag(Wt, WXR, cF, r_off);
    bf16x16 bG = load_b_frag(Wt, WXR, cG, r_off);
    bf16x16 bO = load_b_frag(Wt, WXR, cO, r_off);
    for (int k0 = 0; k0 < HID; k0 += 32) {
      bf16x16 a0 = load_a_frag(hread, HID, 0,  k0);
      bf16x16 a1 = load_a_frag(hread, HID, 16, k0);
      bf16x16 nI = bI, nF = bF, nG = bG, nO = bO;
      if (k0 + 32 < HID) {
        nI = load_b_frag(Wt, WXR, cI, r_off + k0 + 32);
        nF = load_b_frag(Wt, WXR, cF, r_off + k0 + 32);
        nG = load_b_frag(Wt, WXR, cG, r_off + k0 + 32);
        nO = load_b_frag(Wt, WXR, cO, r_off + k0 + 32);
      }
      aI0 = wmma_bf16(a0, bI, aI0);  aI1 = wmma_bf16(a1, bI, aI1);
      aF0 = wmma_bf16(a0, bF, aF0);  aF1 = wmma_bf16(a1, bF, aF1);
      aG0 = wmma_bf16(a0, bG, aG0);  aG1 = wmma_bf16(a1, bG, aG1);
      aO0 = wmma_bf16(a0, bO, aO0);  aO1 = wmma_bf16(a1, bO, aO1);
      bI = nI; bF = nF; bG = nG; bO = nO;
    }

    // elementwise cell update, both m-tiles
#pragma unroll
    for (int v = 0; v < 8; ++v) {
      int kc = ncol + r;
      {
        int b = 8 * half + v;
        size_t ci = (size_t)b * HID + kc;
        float c = sigm(aF0[v]) * cd[ci] + sigm(aI0[v]) * tanh_fast(aG0[v]);
        cd[ci] = c;
        float h = sigm(aO0[v]) * tanh_fast(c);
        out[((size_t)t * NBAT + b) * (2 * HID) + (size_t)dir * HID + kc] = h;
        hwrite[ci] = f2bf(h);
      }
      {
        int b = 16 + 8 * half + v;
        size_t ci = (size_t)b * HID + kc;
        float c = sigm(aF1[v]) * cd[ci] + sigm(aI1[v]) * tanh_fast(aG1[v]);
        cd[ci] = c;
        float h = sigm(aO1[v]) * tanh_fast(c);
        out[((size_t)t * NBAT + b) * (2 * HID) + (size_t)dir * HID + kc] = h;
        hwrite[ci] = f2bf(h);
      }
    }

    // per-direction grid barrier between dependent timesteps
    __syncthreads();
    if (tid == 0) {
      __threadfence();
      atomicAdd(syncd, 1u);
      unsigned target = (unsigned)(NBLK * (s + 1));
      while (atomicAdd(syncd, 0u) < target) { __builtin_amdgcn_s_sleep(1); }
    }
    __syncthreads();
  }
}

// ------------------------------- launcher ----------------------------------

extern "C" void kernel_launch(void* const* d_in, const int* in_sizes, int n_in,
                              void* d_out, int out_size, void* d_ws, size_t ws_size,
                              hipStream_t stream) {
  const float* x  = (const float*)d_in[0];   // (T, B, IN0)
  const float* W  = (const float*)d_in[1];   // (1, WXR, WYC)
  const float* h0 = (const float*)d_in[2];   // (1, 4, B, H)
  const float* c0 = (const float*)d_in[3];   // (1, 4, B, H)
  float* out = (float*)d_out;                // (T, B, 2H)

  char* ws = (char*)d_ws;
  size_t off = 0;
  unsigned short* Wt   = (unsigned short*)(ws + off); off += (size_t)WYC * WXR * 2;
  unsigned short* xbf  = (unsigned short*)(ws + off); off += (size_t)T_SEQ * NBAT * IN0 * 2;
  float*          xw   = (float*)(ws + off);          off += (size_t)T_SEQ * NBAT * WYC * 4;
  float*          out0 = (float*)(ws + off);          off += (size_t)T_SEQ * NBAT * 2 * HID * 4;
  unsigned short* hbf  = (unsigned short*)(ws + off); off += (size_t)4 * 32768 * 2;
  float*          cst  = (float*)(ws + off);          off += (size_t)2 * 32768 * 4;
  unsigned*       sync = (unsigned*)(ws + off);       off += 256;
  (void)ws_size; (void)in_sizes; (void)n_in; (void)out_size;

  transpose_cvt_w<<<dim3(WXR / 32, WYC / 8), dim3(32, 8), 0, stream>>>(W, Wt);

  // ---- layer 0 ----
  int nx0 = T_SEQ * NBAT * IN0;
  cvt_f32_bf16<<<(nx0 + 255) / 256, 256, 0, stream>>>(x, xbf, nx0);
  wmma_gemm_proj<<<dim3(WYC / 256, (T_SEQ * NBAT) / 32), 256, 0, stream>>>(xbf, Wt, xw, IN0, 0);
  init_state<<<256, 256, 0, stream>>>(h0, c0, hbf, cst, sync, 0);
  lstm_scan<<<dim3(NBLK, 2), 256, 0, stream>>>(Wt, xw, out0, hbf, cst, sync, IN0);

  // ---- layer 1 ----
  int nx1 = T_SEQ * NBAT * 2 * HID;
  cvt_f32_bf16<<<(nx1 + 255) / 256, 256, 0, stream>>>(out0, xbf, nx1);
  wmma_gemm_proj<<<dim3(WYC / 256, (T_SEQ * NBAT) / 32), 256, 0, stream>>>(xbf, Wt, xw, 2 * HID, IN0 + HID);
  init_state<<<256, 256, 0, stream>>>(h0, c0, hbf, cst, sync, 1);
  lstm_scan<<<dim3(NBLK, 2), 256, 0, stream>>>(Wt, xw, out, hbf, cst, sync, IN0 + 3 * HID);
}
